// SAGE_24300924961370
// MI455X (gfx1250) — matrix operand order repaired and checked by reading, not compiled, not run on hardware
//
#include <hip/hip_runtime.h>
#include <hip/hip_bf16.h>

// ---------------------------------------------------------------------------
// GraphSAGE 2-layer forward for MI455X (gfx1250, wave32, WMMA).
// N=50000, E=800000, F_IN=H=128, C=64. All GEMMs fused to K=256 bf16 WMMA.
// ---------------------------------------------------------------------------

typedef __attribute__((ext_vector_type(16))) __bf16 v16bf;
typedef __attribute__((ext_vector_type(8)))  __bf16 v8bf;
typedef __attribute__((ext_vector_type(8)))  float  v8f;

#define FDIM 128   // feature width of every activation half (F_IN == H == 128)
#define KCAT 256   // concatenated K for all GEMMs

// --------------------------- degree count ----------------------------------
__global__ void degree_kernel(const int* __restrict__ dst, float* __restrict__ cnt, int E) {
    int e = blockIdx.x * blockDim.x + threadIdx.x;
    if (e < E) atomicAdd(cnt + dst[e], 1.0f);
}

// --------------------- edge scatter-add (one wave / edge) ------------------
// lane L handles features [4L, 4L+4): float4 gather from feat[src], 4 atomic
// f32 adds into accum[dst]. accum (25.6MB) lives in L2 -> atomics stay on-die.
__global__ void scatter_kernel(const float* __restrict__ feat,
                               const int* __restrict__ src,
                               const int* __restrict__ dst,
                               float* __restrict__ accum, int E) {
    int wid  = (int)((blockIdx.x * (unsigned)blockDim.x + threadIdx.x) >> 5);
    int lane = threadIdx.x & 31;
    if (wid >= E) return;
    int s = src[wid];           // uniform across wave -> scalar broadcast
    int d = dst[wid];
    const float4 v = *(const float4*)(feat + (size_t)s * FDIM + lane * 4);
    float* o = accum + (size_t)d * FDIM + lane * 4;
    atomicAdd(o + 0, v.x);
    atomicAdd(o + 1, v.y);
    atomicAdd(o + 2, v.z);
    atomicAdd(o + 3, v.w);
}

// -------- build bf16 concat activation  Abf[n] = [A0[n]*scale | A1[n]] -----
__global__ void prep_cat_kernel(const float* __restrict__ A0,
                                const float* __restrict__ cnt,  // null => scale 1
                                const float* __restrict__ A1,
                                __bf16* __restrict__ out, int n) {
    int i = blockIdx.x * blockDim.x + threadIdx.x;
    if (i >= n * FDIM) return;
    int row = i >> 7, col = i & (FDIM - 1);
    float scale = 1.0f;
    if (cnt) scale = 1.0f / fmaxf(cnt[row], 1.0f);     // mean aggr, deg-0 -> 0
    out[(size_t)row * KCAT + col]        = (__bf16)(A0[i] * scale);
    out[(size_t)row * KCAT + FDIM + col] = (__bf16)A1[i];
}

// --------- build bf16 concat weight  Wbf[o] = [Wa[o,:128] | Wb[o,:128]] ----
__global__ void prep_w_kernel(const float* __restrict__ Wa, const float* __restrict__ Wb,
                              int lda, int ldb, __bf16* __restrict__ out, int Co) {
    int i = blockIdx.x * blockDim.x + threadIdx.x;
    if (i >= Co * FDIM) return;
    int r = i >> 7, c = i & (FDIM - 1);
    out[(size_t)r * KCAT + c]        = (__bf16)Wa[(size_t)r * lda + c];
    out[(size_t)r * KCAT + FDIM + c] = (__bf16)Wb[(size_t)r * ldb + c];
}

// ------------------- fused WMMA GEMM + bias (+L2norm)(+relu) ---------------
// out[M][NT*16] = post( Abf[M][256] @ Wbf[NT*16][256]^T + bias )
// Block: 256 thr = 8 waves; each wave owns a 16-row m-tile x full Co.
// Wbf staged in dynamic LDS (NT*8KB), reused by all waves/k-steps/n-tiles.
// Per k-step: all NT B-frags loaded as one DS clause, then NT back-to-back
// WMMAs into independent accumulators (single s_wait, max XDL co-execution).
template <int NT, bool NORM, bool RELU>
__global__ void gemm_wmma_kernel(const __bf16* __restrict__ A,
                                 const __bf16* __restrict__ W,
                                 const float* __restrict__ bias,
                                 float* __restrict__ out, int M) {
    extern __shared__ __bf16 wlds[];                 // [NT*16][256]
    const int Co = NT * 16;

    // cooperative 16B-vector copy of W into LDS
    for (int i = threadIdx.x; i < Co * (KCAT / 8); i += blockDim.x)
        ((v8bf*)wlds)[i] = ((const v8bf*)W)[i];
    __syncthreads();

    const int wave  = threadIdx.x >> 5;
    const int lane  = threadIdx.x & 31;
    const int half  = lane >> 4;                     // 0: lanes 0-15, 1: 16-31
    const int l16   = lane & 15;
    const int mbase = blockIdx.x * 128 + wave * 16;
    if (mbase >= M) return;                          // wave-uniform

    const int m = mbase + l16;
    const __bf16* arow = A + (size_t)(m < M ? m : 0) * KCAT;
    const __bf16* wbase = wlds + (size_t)l16 * KCAT + half * 16;

    v8f acc[NT];
#pragma unroll
    for (int t = 0; t < NT; ++t) acc[t] = (v8f){0.f, 0.f, 0.f, 0.f, 0.f, 0.f, 0.f, 0.f};

#pragma unroll
    for (int k = 0; k < KCAT; k += 32) {
        // A-frag (16x32 bf16): lane-half h holds K [k+16h, k+16h+16)
        v8bf alo = *(const v8bf*)(arow + k + half * 16);
        v8bf ahi = *(const v8bf*)(arow + k + half * 16 + 8);
        v16bf afrag = __builtin_shufflevector(alo, ahi,
                        0, 1, 2, 3, 4, 5, 6, 7, 8, 9, 10, 11, 12, 13, 14, 15);

        // B-frags (32x16 bf16): lane = col n, half h holds K [k+16h, k+16h+16).
        // Load ALL tiles first -> one DS clause + one wait.
        v16bf bfrag[NT];
#pragma unroll
        for (int t = 0; t < NT; ++t) {
            const __bf16* wrow = wbase + (size_t)t * 16 * KCAT + k;
            v8bf blo = *(const v8bf*)(wrow);
            v8bf bhi = *(const v8bf*)(wrow + 8);
            bfrag[t] = __builtin_shufflevector(blo, bhi,
                         0, 1, 2, 3, 4, 5, 6, 7, 8, 9, 10, 11, 12, 13, 14, 15);
        }
#pragma unroll
        for (int t = 0; t < NT; ++t)
            acc[t] = __builtin_amdgcn_wmma_f32_16x16x32_bf16(
                false, afrag, false, bfrag[t], (short)0, acc[t], false, false);
    }

    // epilogue: bias -> (L2 row norm) -> (relu)
#pragma unroll
    for (int t = 0; t < NT; ++t) {
        float bv = bias[t * 16 + l16];
#pragma unroll
        for (int r = 0; r < 8; ++r) acc[t][r] += bv;
    }

    if (NORM) {
        // C/D layout: element r, lane -> row mbase + r + 8*half, col t*16+l16.
        // Each output row lives in one 16-lane half-wave -> shfl_xor width 16.
#pragma unroll
        for (int r = 0; r < 8; ++r) {
            float s = 0.f;
#pragma unroll
            for (int t = 0; t < NT; ++t) s += acc[t][r] * acc[t][r];
            s += __shfl_xor(s, 1, 16);
            s += __shfl_xor(s, 2, 16);
            s += __shfl_xor(s, 4, 16);
            s += __shfl_xor(s, 8, 16);
            float inv = 1.0f / fmaxf(sqrtf(s), 1e-12f);
#pragma unroll
            for (int t = 0; t < NT; ++t) acc[t][r] *= inv;
        }
    }
    if (RELU) {
#pragma unroll
        for (int t = 0; t < NT; ++t)
#pragma unroll
            for (int r = 0; r < 8; ++r) acc[t][r] = fmaxf(acc[t][r], 0.f);
    }

#pragma unroll
    for (int r = 0; r < 8; ++r) {
        int mr = mbase + r + half * 8;
        if (mr < M) {
#pragma unroll
            for (int t = 0; t < NT; ++t)
                out[(size_t)mr * Co + t * 16 + l16] = acc[t][r];
        }
    }
}

// ---------------------------------------------------------------------------
extern "C" void kernel_launch(void* const* d_in, const int* in_sizes, int n_in,
                              void* d_out, int out_size, void* d_ws, size_t ws_size,
                              hipStream_t stream) {
    const float* x    = (const float*)d_in[0];
    const int*   ei   = (const int*)d_in[1];
    const float* W1_l = (const float*)d_in[2];
    const float* b1_l = (const float*)d_in[3];
    const float* W1_r = (const float*)d_in[4];
    const float* Wl1  = (const float*)d_in[5];
    const float* bl1  = (const float*)d_in[6];
    const float* W2_l = (const float*)d_in[7];
    const float* b2_l = (const float*)d_in[8];
    const float* W2_r = (const float*)d_in[9];

    const int N = in_sizes[0] / FDIM;
    const int E = in_sizes[1] / 2;
    const int H = 128, C = 64;
    const int* src = ei;
    const int* dst = ei + E;

    // workspace layout (256B aligned)
    char* p = (char*)d_ws;
    auto take = [&](size_t bytes) { char* r = p; p += (bytes + 255) & ~(size_t)255; return r; };
    float*  cnt = (float*)take((size_t)N * 4);
    float*  agg = (float*)take((size_t)N * FDIM * 4);
    float*  h1  = (float*)take((size_t)N * H * 4);
    float*  h2  = (float*)take((size_t)N * H * 4);
    __bf16* Abf = (__bf16*)take((size_t)N * KCAT * 2);
    __bf16* Wbf = (__bf16*)take((size_t)H * KCAT * 2);

    const int T = 256;
    const int gemmGrid  = (N + 127) / 128;
    const int elemGrid  = (N * FDIM + T - 1) / T;
    const int edgeGrid  = (E + T - 1) / T;
    const int waveGrid  = (E * 32 + T - 1) / T;   // one wave per edge
    const int w128Grid  = (H * FDIM + T - 1) / T;
    const int w64Grid   = (C * FDIM + T - 1) / T;

    // ---- layer 1: h1 = relu(l2norm([mean_aggr(x) | x] @ [W1_l|W1_r]^T + b1))
    hipMemsetAsync(cnt, 0, (size_t)N * 4, stream);
    hipMemsetAsync(agg, 0, (size_t)N * FDIM * 4, stream);
    degree_kernel<<<edgeGrid, T, 0, stream>>>(dst, cnt, E);
    scatter_kernel<<<waveGrid, T, 0, stream>>>(x, src, dst, agg, E);
    prep_cat_kernel<<<elemGrid, T, 0, stream>>>(agg, cnt, x, Abf, N);
    prep_w_kernel<<<w128Grid, T, 0, stream>>>(W1_l, W1_r, FDIM, FDIM, Wbf, H);
    gemm_wmma_kernel<8, true, true><<<gemmGrid, T, H * KCAT * 2, stream>>>(Abf, Wbf, b1_l, h1, N);

    // ---- middle MLP: h2 = relu([x | h1] @ Wl1^T + bl1)
    prep_cat_kernel<<<elemGrid, T, 0, stream>>>(x, nullptr, h1, Abf, N);
    prep_w_kernel<<<w128Grid, T, 0, stream>>>(Wl1, Wl1 + FDIM, KCAT, KCAT, Wbf, H);
    gemm_wmma_kernel<8, false, true><<<gemmGrid, T, H * KCAT * 2, stream>>>(Abf, Wbf, bl1, h2, N);

    // ---- layer 2: out = l2norm([mean_aggr(h2) | h2] @ [W2_l|W2_r]^T + b2)
    hipMemsetAsync(agg, 0, (size_t)N * FDIM * 4, stream);
    scatter_kernel<<<waveGrid, T, 0, stream>>>(h2, src, dst, agg, E);
    prep_cat_kernel<<<elemGrid, T, 0, stream>>>(agg, cnt, h2, Abf, N);
    prep_w_kernel<<<w64Grid, T, 0, stream>>>(W2_l, W2_r, FDIM, FDIM, Wbf, C);
    gemm_wmma_kernel<4, true, false><<<gemmGrid, T, C * KCAT * 2, stream>>>(Abf, Wbf, b2_l, (float*)d_out, N);
}
